// ConvOffset2d_63909113364759
// MI455X (gfx1250) — compile-verified, matrix operand order checked by reading, not run on hardware
//
#include <hip/hip_runtime.h>
#include <hip/hip_bf16.h>

typedef __attribute__((ext_vector_type(16))) _Float16 v16h;
typedef __attribute__((ext_vector_type(8)))  _Float16 v8h;
typedef __attribute__((ext_vector_type(8)))  float    v8f;

#define CIN   128
#define COUT  128
#define HH    64
#define WW    64
#define TAPS  9
#define KTOT  (CIN * TAPS)      // 1152
#define NSTEP (KTOT / 32)       // 36 K-steps of 32
#define HW    (HH * WW)         // 4096

// ---------------------------------------------------------------------------
// Kernel 1: weights f32 -> f16, pre-swizzled into the WMMA A-operand
// (16x32 f16) lane layout: flat = s*4096 + m*32 + laneHi*16 + h holds
// W[m, k], k = s*32 + h + 8*(h>=8) + 8*laneHi, with k = tap*128 + c.
// ---------------------------------------------------------------------------
__global__ __launch_bounds__(256) void swizzle_weights_f16(
    const float* __restrict__ wgt, _Float16* __restrict__ wA) {
  int tid = blockIdx.x * 256 + threadIdx.x;
  if (tid >= NSTEP * COUT * 32) return;
  int s      = tid >> 12;
  int m      = (tid >> 5) & 127;
  int laneHi = (tid >> 4) & 1;
  int h      = tid & 15;
  int kk = h + ((h >= 8) ? 8 : 0) + laneHi * 8;
  int k  = s * 32 + kk;
  int c   = k & 127;
  int tap = k >> 7;
  wA[tid] = (_Float16)wgt[(m * CIN + c) * TAPS + tap];
}

// ---------------------------------------------------------------------------
// Kernel 2: NCHW f32 -> NHWC f16 staging of x, via LDS tile.
// Block = one (b, y) row: reads coalesced along x, writes coalesced along c.
// ---------------------------------------------------------------------------
__global__ __launch_bounds__(256) void nchw_to_nhwc_f16(
    const float* __restrict__ x, _Float16* __restrict__ xT) {
  __shared__ _Float16 tile[64 * 136];     // pad keeps 16B-aligned v8h reads
  const int by  = blockIdx.x;             // b*64 + y
  const int b   = by >> 6, y = by & 63;
  const int tid = threadIdx.x;
  const int xc  = tid & 63;               // x column
  const int c0  = tid >> 6;               // 0..3

  const float* src = x + (size_t)b * (CIN * HW) + (size_t)y * WW + xc;
#pragma unroll
  for (int i = 0; i < 32; ++i) {
    const int c = i * 4 + c0;
    tile[xc * 136 + c] = (_Float16)src[(size_t)c * HW];
  }
  __syncthreads();

  _Float16* dst = xT + (size_t)by * (WW * CIN) + (size_t)tid * 32;
  const int xo = tid >> 2, co = (tid & 3) * 32;
#pragma unroll
  for (int j = 0; j < 4; ++j)
    *(v8h*)(dst + j * 8) = *(const v8h*)(tile + xo * 136 + co + j * 8);
}

// ---------------------------------------------------------------------------
// Kernel 3: fused deformable-im2col + WMMA GEMM.
// Block = 256 threads = 8 waves, covers one output row (b, ho) x all Cout.
// Wave w: Cout rows [16w,16w+16), 4 N-subtiles of 16 pixels.
// Double-buffered 32(K) x 64(N) f16 column tile in LDS; 1 barrier/step.
// Interpolation runs in packed f16 (v_pk_fma_f16) and co-executes with WMMA.
// ---------------------------------------------------------------------------
__global__ __launch_bounds__(256) void deform_conv2d_wmma(
    const _Float16* __restrict__ xT, const float* __restrict__ off,
    const _Float16* __restrict__ wA, float* __restrict__ out) {
  __shared__ __align__(32) _Float16 colsLDS[2][64 * 32];   // 2 x 4 KB

  const int n0   = blockIdx.x * 64;
  const int b    = n0 >> 12;
  const int ho   = (n0 >> 6) & 63;
  const int tid  = threadIdx.x;
  const int lane = tid & 31;
  const int wv   = tid >> 5;              // wave id -> m0 = 16*wv
  const int p    = tid & 63;              // pixel (== wo)
  const int g    = tid >> 6;              // channel-group 0..3 (8 ch each)

  v8f acc0 = {}, acc1 = {}, acc2 = {}, acc3 = {};

  const float fy0 = (float)(ho - 1);
  const float fx0 = (float)(p - 1);
  const float* offB = off + (size_t)b * (2 * TAPS * HW) + ho * WW + p;
  const _Float16* xTb = xT + (size_t)b * (HW * CIN);

  const _Float16* aLane =
      wA + (size_t)(wv * 16 + (lane & 15)) * 32 + (size_t)(lane >> 4) * 16;
  const int ldsBase = p * 32 + (g & 1) * 16 + (g >> 1) * 8;  // B-layout slot
  const int bOff    = (lane & 15) * 32 + (lane >> 4) * 16;

  for (int s = 0; s < NSTEP; ++s) {
    // ---- deformable sampling coords (shared by 32 channels) --------------
    const int tap  = s >> 2;
    const int cblk = (s & 3) * 32;
    const int ky = tap / 3, kx = tap - 3 * ky;

    const float dy = offB[(size_t)(2 * tap) * HW];
    const float dx = offB[(size_t)(2 * tap + 1) * HW];
    const float py = fy0 + (float)ky + dy;
    const float px = fx0 + (float)kx + dx;
    const float fy = floorf(py), fx = floorf(px);
    const int y0 = (int)fy, x0 = (int)fx;
    const float wy = py - fy, wx = px - fx;

    const bool vy0 = (y0 >= 0) && (y0 < HH);
    const bool vy1 = (y0 + 1 >= 0) && (y0 + 1 < HH);
    const bool vx0 = (x0 >= 0) && (x0 < WW);
    const bool vx1 = (x0 + 1 >= 0) && (x0 + 1 < WW);
    // fold validity into the bilinear weights -> unconditional clamped loads
    const _Float16 w00 = (_Float16)((vy0 && vx0) ? (1.f - wy) * (1.f - wx) : 0.f);
    const _Float16 w01 = (_Float16)((vy0 && vx1) ? (1.f - wy) * wx : 0.f);
    const _Float16 w10 = (_Float16)((vy1 && vx0) ? wy * (1.f - wx) : 0.f);
    const _Float16 w11 = (_Float16)((vy1 && vx1) ? wy * wx : 0.f);

    const int cy0 = min(max(y0, 0), HH - 1);
    const int cy1 = min(max(y0 + 1, 0), HH - 1);
    const int cx0 = min(max(x0, 0), WW - 1);
    const int cx1 = min(max(x0 + 1, 0), WW - 1);

    // ---- gather: 4 corners x 8 contiguous f16 channels (b128 each) -------
    const int cb = cblk + g * 8;
    const _Float16* r0 = xTb + (size_t)(cy0 * WW) * CIN + cb;
    const _Float16* r1 = xTb + (size_t)(cy1 * WW) * CIN + cb;
    const v8h a00 = *(const v8h*)(r0 + (size_t)cx0 * CIN);
    const v8h a01 = *(const v8h*)(r0 + (size_t)cx1 * CIN);
    const v8h a10 = *(const v8h*)(r1 + (size_t)cx0 * CIN);
    const v8h a11 = *(const v8h*)(r1 + (size_t)cx1 * CIN);

    // packed f16 bilinear: 16 v_pk_fma/mul_f16 for 8 channels
    v8h res = a00 * w00;
    res += a01 * w01;
    res += a10 * w10;
    res += a11 * w11;

    _Float16* buf = colsLDS[s & 1];
    *(v8h*)(buf + ldsBase) = res;                 // single ds_store_b128

    if (s + 1 < NSTEP)
      __builtin_prefetch(aLane + (size_t)(s + 1) * 4096, 0, 3);

    __syncthreads();   // producers done; consumers of other buffer finished

    // ---- 4 x v_wmma_f32_16x16x32_f16 per wave ----------------------------
    const v16h a  = *(const v16h*)(aLane + (size_t)s * 4096);
    const v16h b0 = *(const v16h*)(buf + 0 * 512 + bOff);
    const v16h b1 = *(const v16h*)(buf + 1 * 512 + bOff);
    const v16h b2 = *(const v16h*)(buf + 2 * 512 + bOff);
    const v16h b3 = *(const v16h*)(buf + 3 * 512 + bOff);

    acc0 = __builtin_amdgcn_wmma_f32_16x16x32_f16(false, a, false, b0,
                                                  (short)0, acc0, false, false);
    acc1 = __builtin_amdgcn_wmma_f32_16x16x32_f16(false, a, false, b1,
                                                  (short)0, acc1, false, false);
    acc2 = __builtin_amdgcn_wmma_f32_16x16x32_f16(false, a, false, b2,
                                                  (short)0, acc2, false, false);
    acc3 = __builtin_amdgcn_wmma_f32_16x16x32_f16(false, a, false, b3,
                                                  (short)0, acc3, false, false);
    // no trailing barrier: next step writes the other buffer
  }

  // ---- epilogue: f32 C/D layout -> out[b, m, ho, wo] ----------------------
  const int nlo = lane & 15;
  const int mhi = (lane >> 4) * 8;
  float* outRow = out + (size_t)b * COUT * HW + ho * WW;
#pragma unroll
  for (int r = 0; r < 8; ++r) {
    const int m = wv * 16 + mhi + r;
    float* o = outRow + (size_t)m * HW;
    o[0 * 16 + nlo] = acc0[r];
    o[1 * 16 + nlo] = acc1[r];
    o[2 * 16 + nlo] = acc2[r];
    o[3 * 16 + nlo] = acc3[r];
  }
}

// ---------------------------------------------------------------------------
extern "C" void kernel_launch(void* const* d_in, const int* in_sizes, int n_in,
                              void* d_out, int out_size, void* d_ws,
                              size_t ws_size, hipStream_t stream) {
  (void)in_sizes; (void)n_in; (void)out_size; (void)ws_size;
  const float* x   = (const float*)d_in[0];
  const float* off = (const float*)d_in[1];
  const float* wgt = (const float*)d_in[2];
  float* out = (float*)d_out;

  _Float16* wA = (_Float16*)d_ws;                 // 294912 B
  _Float16* xT = wA + (size_t)NSTEP * COUT * 32;  // + 8 MB NHWC f16 x

  swizzle_weights_f16<<<(NSTEP * COUT * 32 + 255) / 256, 256, 0, stream>>>(wgt, wA);
  nchw_to_nhwc_f16<<<8 * HH, 256, 0, stream>>>(x, xT);
  deform_conv2d_wmma<<<(8 * HW) / 64, 256, 0, stream>>>(xT, off, wA, out);
}
